// AttentionMechanism_8641474200006
// MI455X (gfx1250) — matrix operand order repaired and checked by reading, not compiled, not run on hardware
//
#include <hip/hip_runtime.h>

typedef __attribute__((ext_vector_type(16))) __bf16 v16bf;
typedef __attribute__((ext_vector_type(8)))  float  v8f;

// Problem constants (from reference setup_inputs)
static constexpr int NB   = 64;     // batch
static constexpr int NT   = 1500;   // time
static constexpr int ND   = 1024;   // enc dim
static constexpr int NA   = 512;    // attention dim
static constexpr int NC   = 10;     // conv channels
static constexpr int NKW  = 101;    // conv kernel width
static constexpr int NPAD = 50;
static constexpr int BT   = NB * NT;        // 96000 rows
static constexpr int KK   = ND + 32;        // padded GEMM K (1024 enc + 10 conv + pad) = 1056
static constexpr int KSTEPS = KK / 32;      // 33

// ---------------------------------------------------------------------------
// Prep 1: transpose+convert weights to bf16 B-matrix layout Wt[a][k], k-major.
//   k <  1024  : W_w[k][a]         (enc half of the concat projection)
//   1024..1033 : Wc_w[k-1024][a]   (conv-feature projection folded into K)
//   else       : 0
// ---------------------------------------------------------------------------
__global__ void prep_wt(const float* __restrict__ W_w,
                        const float* __restrict__ Wc_w,
                        __bf16* __restrict__ Wt) {
    int idx = blockIdx.x * blockDim.x + threadIdx.x;
    if (idx >= NA * KK) return;
    int a = idx / KK;
    int k = idx % KK;
    float v = 0.0f;
    if (k < ND)            v = W_w[k * NA + a];
    else if (k < ND + NC)  v = Wc_w[(k - ND) * NA + a];
    Wt[a * KK + k] = (__bf16)v;
}

// ---------------------------------------------------------------------------
// Prep 2: per-batch decoder projection + bias: decb[b][a] = W_b[a] + dec·W_w[D:,a]
// ---------------------------------------------------------------------------
__global__ void prep_decb(const float* __restrict__ dec_out,
                          const float* __restrict__ W_w,
                          const float* __restrict__ W_b,
                          float* __restrict__ decb) {
    int b = blockIdx.x;
    for (int a = threadIdx.x; a < NA; a += blockDim.x) {
        float acc = W_b[a];
        const float* wcol = W_w + (size_t)ND * NA + a;   // rows D..2D-1, column a
        const float* dv   = dec_out + (size_t)b * ND;
        for (int d = 0; d < ND; ++d) acc += dv[d] * wcol[(size_t)d * NA];
        decb[b * NA + a] = acc;
    }
}

// ---------------------------------------------------------------------------
// Prep 3: 1D conv of previous attention weights -> padded A-matrix tail.
//   convA[row][c] = sum_k att[b][t+k-PAD] * conv_k[c][0][k], c<10; zero-pad to 32.
// ---------------------------------------------------------------------------
__global__ void prep_conv(const float* __restrict__ attw,
                          const float* __restrict__ convk,
                          float* __restrict__ convA) {
    int row = blockIdx.x * blockDim.x + threadIdx.x;
    if (row >= BT) return;
    int b = row / NT, t = row % NT;
    const float* ab = attw + (size_t)b * NT;
    float o[NC];
#pragma unroll
    for (int c = 0; c < NC; ++c) o[c] = 0.0f;
    for (int k = 0; k < NKW; ++k) {
        int ti = t + k - NPAD;
        if (ti < 0 || ti >= NT) continue;
        float x = ab[ti];
#pragma unroll
        for (int c = 0; c < NC; ++c) o[c] += x * convk[c * NKW + k];
    }
    float* dst = convA + (size_t)row * 32;
#pragma unroll
    for (int i = 0; i < 32; ++i) dst[i] = (i < NC) ? o[i] : 0.0f;
}

// ---------------------------------------------------------------------------
// Main fused kernel: bf16 WMMA GEMM (hidden = A @ Wt^T), + dec/bias, tanh, ·V,
// reduce over A -> raw energy[row].
// Block: 256 threads = 8 waves. Block tile: 32 rows x 512 cols, K = 1056.
// Wave w: m-half = w>>2, n-quarter (128 cols = 8 WMMA tiles) = w&3.
// ---------------------------------------------------------------------------
__global__ __launch_bounds__(256)
void energy_gemm(const float*  __restrict__ enc,
                 const float*  __restrict__ convA,
                 const __bf16* __restrict__ Wt,
                 const float*  __restrict__ decb,
                 const float*  __restrict__ Vw,
                 float* __restrict__ energy) {
    __shared__ float red[4 * 16 * 32];   // [n_quarter][lane%16][row_local]

    const int tid  = threadIdx.x;
    const int wv   = tid >> 5;
    const int lane = tid & 31;
    const int mh   = wv >> 2;          // 0/1: which 16-row half
    const int nq   = wv & 3;           // 0..3: 128-column quarter
    const int l16  = lane & 15;
    const bool hi  = lane >= 16;

    const int rowA = blockIdx.x * 32 + mh * 16 + l16;   // A row this lane feeds

    v8f acc[8];
#pragma unroll
    for (int j = 0; j < 8; ++j) acc[j] = (v8f){0.f,0.f,0.f,0.f,0.f,0.f,0.f,0.f};

    // A sources (f32, converted to bf16 in-register). Per the 16-bit A layout:
    // lane<16 holds K {k0..k0+7, k0+16..k0+23}; lane>=16 holds {+8, +24}.
    const float* arow_enc = enc   + (size_t)rowA * ND + (hi ? 8 : 0);
    const float* arow_cnv = convA + (size_t)rowA * 32 + (hi ? 8 : 0);
    // B source: Wt[a][k] contiguous in k. Lane<16: K {k0..k0+7, k0+16..k0+23}; lane>=16: +8.
    const __bf16* wbase = Wt + (size_t)(nq * 128 + l16) * KK + (hi ? 8 : 0);

    for (int ks = 0; ks < KSTEPS; ++ks) {
        const float* ap = (ks < KSTEPS - 1) ? (arow_enc + ks * 32) : arow_cnv;
        const float4* ap4 = (const float4*)ap;
        float4 f0 = ap4[0], f1 = ap4[1];   // K k0 .. k0+7   (this lane's half)
        float4 f2 = ap4[4], f3 = ap4[5];   // K k0+16 .. k0+23
        v16bf af;
        af[0]=(__bf16)f0.x; af[1]=(__bf16)f0.y; af[2]=(__bf16)f0.z; af[3]=(__bf16)f0.w;
        af[4]=(__bf16)f1.x; af[5]=(__bf16)f1.y; af[6]=(__bf16)f1.z; af[7]=(__bf16)f1.w;
        af[8]=(__bf16)f2.x; af[9]=(__bf16)f2.y; af[10]=(__bf16)f2.z; af[11]=(__bf16)f2.w;
        af[12]=(__bf16)f3.x; af[13]=(__bf16)f3.y; af[14]=(__bf16)f3.z; af[15]=(__bf16)f3.w;
#pragma unroll
        for (int j = 0; j < 8; ++j) {
            const __bf16* bp = wbase + (size_t)j * 16 * KK + ks * 32;
            v16bf bfr;
#pragma unroll
            for (int s = 0; s < 8; ++s) { bfr[s] = bp[s]; bfr[8 + s] = bp[16 + s]; }
            acc[j] = __builtin_amdgcn_wmma_f32_16x16x32_bf16(
                false, af, false, bfr, (short)0, acc[j], false, false);
        }
    }

    // Epilogue: D layout -> element (vgpr r, lane) is row (r + 8*hi), col l16.
    const int grb = blockIdx.x * 32 + mh * 16 + (hi ? 8 : 0);
    int bbat[8];
#pragma unroll
    for (int r = 0; r < 8; ++r) bbat[r] = (grb + r) / NT;

    float psum[8];
#pragma unroll
    for (int r = 0; r < 8; ++r) psum[r] = 0.0f;

#pragma unroll
    for (int j = 0; j < 8; ++j) {
        int n = nq * 128 + j * 16 + l16;
        float vw = Vw[n];
#pragma unroll
        for (int r = 0; r < 8; ++r) {
            float h = acc[j][r] + decb[bbat[r] * NA + n];
            psum[r] += vw * tanhf(h);
        }
    }

    // Deterministic cross-wave reduction via LDS (no atomics).
#pragma unroll
    for (int r = 0; r < 8; ++r) {
        int rl = mh * 16 + r + (hi ? 8 : 0);
        red[(nq * 16 + l16) * 32 + rl] = psum[r];
    }
    __syncthreads();
    if (tid < 32) {
        float s = 0.0f;
        for (int i = 0; i < 64; ++i) s += red[i * 32 + tid];
        energy[blockIdx.x * 32 + tid] = 2.0f * s;   // SHARPENING_FACTOR
    }
}

// ---------------------------------------------------------------------------
// Softmax over T, in place on the att region of d_out. One block per batch.
// ---------------------------------------------------------------------------
__global__ void softmax_k(float* __restrict__ att) {
    __shared__ float sd[256];
    const int b = blockIdx.x, tid = threadIdx.x;
    float* e = att + (size_t)b * NT;
    float m = -1e30f;
    for (int t = tid; t < NT; t += 256) m = fmaxf(m, e[t]);
    sd[tid] = m; __syncthreads();
    for (int s = 128; s > 0; s >>= 1) {
        if (tid < s) sd[tid] = fmaxf(sd[tid], sd[tid + s]);
        __syncthreads();
    }
    float mx = sd[0];
    __syncthreads();
    float sum = 0.0f;
    for (int t = tid; t < NT; t += 256) { float v = __expf(e[t] - mx); e[t] = v; sum += v; }
    sd[tid] = sum; __syncthreads();
    for (int s = 128; s > 0; s >>= 1) {
        if (tid < s) sd[tid] += sd[tid + s];
        __syncthreads();
    }
    float inv = 1.0f / sd[0];
    for (int t = tid; t < NT; t += 256) e[t] *= inv;
}

// ---------------------------------------------------------------------------
// Context: ctx[b][d] = sum_t enc[b][t][d] * att[b][t]. Bandwidth-bound matvec.
// Grid (64, 4), block 256: thread owns one d column.
// ---------------------------------------------------------------------------
__global__ void context_k(const float* __restrict__ enc,
                          const float* __restrict__ att,
                          float* __restrict__ ctx) {
    __shared__ float sat[NT];
    const int b  = blockIdx.x;
    const int dd = blockIdx.y * 256 + threadIdx.x;
    for (int t = threadIdx.x; t < NT; t += 256) sat[t] = att[(size_t)b * NT + t];
    __syncthreads();
    const float* eb = enc + (size_t)b * NT * ND + dd;
    float a0 = 0.f, a1 = 0.f, a2 = 0.f, a3 = 0.f;
    int t = 0;
    for (; t + 4 <= NT; t += 4) {
        a0 += eb[(size_t)(t + 0) * ND] * sat[t + 0];
        a1 += eb[(size_t)(t + 1) * ND] * sat[t + 1];
        a2 += eb[(size_t)(t + 2) * ND] * sat[t + 2];
        a3 += eb[(size_t)(t + 3) * ND] * sat[t + 3];
    }
    for (; t < NT; ++t) a0 += eb[(size_t)t * ND] * sat[t];
    ctx[(size_t)b * ND + dd] = a0 + a1 + a2 + a3;
}

// ---------------------------------------------------------------------------
extern "C" void kernel_launch(void* const* d_in, const int* in_sizes, int n_in,
                              void* d_out, int out_size, void* d_ws, size_t ws_size,
                              hipStream_t stream) {
    const float* enc   = (const float*)d_in[0];  // [B,T,D]
    const float* dec   = (const float*)d_in[1];  // [B,1,D]
    const float* attw  = (const float*)d_in[2];  // [B,T]
    const float* W_w   = (const float*)d_in[3];  // [2D,A]
    const float* W_b   = (const float*)d_in[4];  // [A]
    const float* Wc_w  = (const float*)d_in[5];  // [C,A]
    const float* V_w   = (const float*)d_in[6];  // [A,1]
    const float* convk = (const float*)d_in[7];  // [C,1,K]

    float* out = (float*)d_out;
    float* ctx = out;                     // [B,1,D] = 65536 floats
    float* att = out + NB * ND;           // [B,T]   = 96000 floats (energy->softmax in place)

    // Workspace layout (all offsets 256B-aligned):
    char* ws = (char*)d_ws;
    __bf16* Wt   = (__bf16*)(ws);                                     // 512*1056*2 = 1,081,344 B
    float*  decb = (float*)(ws + 1081344);                            // 64*512*4   =   131,072 B
    float*  cnvA = (float*)(ws + 1081344 + 131072);                   // 96000*32*4 = 12,288,000 B

    prep_wt  <<<(NA * KK + 255) / 256, 256, 0, stream>>>(W_w, Wc_w, Wt);
    prep_decb<<<NB, 256, 0, stream>>>(dec, W_w, W_b, decb);
    prep_conv<<<(BT + 255) / 256, 256, 0, stream>>>(attw, convk, cnvA);
    energy_gemm<<<BT / 32, 256, 0, stream>>>(enc, cnvA, Wt, decb, V_w, att);
    softmax_k<<<NB, 256, 0, stream>>>(att);
    context_k<<<dim3(NB, ND / 256), 256, 0, stream>>>(enc, att, ctx);
}
